// Causual_attentionV1_56521769615960
// MI455X (gfx1250) — matrix operand order repaired and checked by reading, not compiled, not run on hardware
//
#include <hip/hip_runtime.h>
#include <hip/hip_bf16.h>

// ---------------------------------------------------------------------------
// Causal attention (B=4, T=2048, D=1024, no head split) for gfx1250 (CDNA5).
// bf16 WMMA (v_wmma_f32_16x16x32_bf16) with f32 accumulation everywhere.
// ---------------------------------------------------------------------------

typedef float  v8f   __attribute__((ext_vector_type(8)));
typedef __bf16 v16bf __attribute__((ext_vector_type(16)));
typedef __bf16 bf8   __attribute__((ext_vector_type(8)));

#define BB 4
#define TT 2048
#define DD 1024
#define MM (BB * TT)   // 8192 rows of x

static __device__ __forceinline__ v16bf make_frag(bf8 lo, bf8 hi) {
    v16bf r;
#pragma unroll
    for (int i = 0; i < 8; ++i) { r[i] = lo[i]; r[i + 8] = hi[i]; }
    return r;
}

static __device__ __forceinline__ v8f wmma_bf16(v16bf a, v16bf b, v8f c) {
    return __builtin_amdgcn_wmma_f32_16x16x32_bf16(false, a, false, b,
                                                   (short)0, c, false, false);
}

// ---------------------------------------------------------------------------
// Kernel 0a: x (f32) -> xb (bf16), elementwise
// ---------------------------------------------------------------------------
__global__ void cvt_x_kernel(const float* __restrict__ x,
                             __bf16* __restrict__ xb, int n) {
    int i = blockIdx.x * blockDim.x + threadIdx.x;
    if (i < n) xb[i] = (__bf16)x[i];
}

// ---------------------------------------------------------------------------
// Kernel 0b: W [K=D_in, N=D_out] f32 -> Wt [N, K] bf16 (transposed), 3 weights
// ---------------------------------------------------------------------------
__global__ void cvt_w_kernel(const float* __restrict__ wq,
                             const float* __restrict__ wk,
                             const float* __restrict__ wv,
                             __bf16* __restrict__ wqt,
                             __bf16* __restrict__ wkt,
                             __bf16* __restrict__ wvt) {
    int z = blockIdx.y;
    const float* w = (z == 0) ? wq : (z == 1) ? wk : wv;
    __bf16* wt     = (z == 0) ? wqt : (z == 1) ? wkt : wvt;
    int i = blockIdx.x * blockDim.x + threadIdx.x;  // i in [0, D*D)
    int k = i >> 10;          // row of W   (d_in)
    int n = i & (DD - 1);     // col of W   (d_out)
    wt[(size_t)n * DD + k] = (__bf16)w[i];
}

// ---------------------------------------------------------------------------
// Kernel 1: QKV projection GEMM.  y = x @ W  via  A = x (16x32 frags),
// B = Wt rows (K-contiguous).  Block = 256 thr (8 waves), tile 128(M) x 64(N),
// wave tile 32x32 -> 4 accumulators.  z in {0,1,2} selects Q / K / V.
// Q stored pre-scaled by 1/sqrt(D).  Q,K row-major bf16 [B*T, D];
// V stored TRANSPOSED bf16 [B, D, T].  Destination branch hoisted out of the
// store loops (block-uniform) so the epilogue is two clean store sequences.
// ---------------------------------------------------------------------------
__global__ void qkv_gemm_kernel(const __bf16* __restrict__ xb,
                                const __bf16* __restrict__ wqt,
                                const __bf16* __restrict__ wkt,
                                const __bf16* __restrict__ wvt,
                                __bf16* __restrict__ qb,
                                __bf16* __restrict__ kbuf,
                                __bf16* __restrict__ vt) {
    const int z = blockIdx.z;
    const __bf16* wt = (z == 0) ? wqt : (z == 1) ? wkt : wvt;

    const int lane = threadIdx.x & 31;
    const int w    = threadIdx.x >> 5;
    const int wm   = w & 3;            // 4 waves along M
    const int wn   = w >> 2;           // 2 waves along N
    const int m0   = blockIdx.y * 128 + wm * 32;
    const int n0   = blockIdx.x * 64  + wn * 32;

    const int lm  = lane & 15;
    const int hiA = (lane >= 16) ? 8 : 0;    // A-fragment K-chunk base
    const int hiB = (lane >= 16) ? 16 : 0;   // unused: B fragments also use hiA

    v8f acc[2][2];
#pragma unroll
    for (int i = 0; i < 2; ++i)
#pragma unroll
        for (int j = 0; j < 2; ++j) acc[i][j] = (v8f)(0.0f);

    for (int k0 = 0; k0 < DD; k0 += 32) {
        v16bf a[2], b[2];
#pragma unroll
        for (int i = 0; i < 2; ++i) {
            const __bf16* pa = xb + (size_t)(m0 + i * 16 + lm) * DD + k0 + hiA;
            a[i] = make_frag(*(const bf8*)pa, *(const bf8*)(pa + 16));
            const __bf16* pb = wt + (size_t)(n0 + i * 16 + lm) * DD + k0 + hiA;
            b[i] = make_frag(*(const bf8*)pb, *(const bf8*)(pb + 16));
        }
#pragma unroll
        for (int i = 0; i < 2; ++i)
#pragma unroll
            for (int j = 0; j < 2; ++j)
                acc[i][j] = wmma_bf16(a[i], b[j], acc[i][j]);
    }

    const int rb = (lane >= 16) ? 8 : 0;     // C/D layout row base
    if (z < 2) {
        // Q pre-scaled by softmax scale 1/sqrt(1024); K unscaled.
        const float mult = (z == 0) ? 0.03125f : 1.0f;
        __bf16* dst = (z == 0) ? qb : kbuf;
#pragma unroll
        for (int i = 0; i < 2; ++i)
#pragma unroll
            for (int j = 0; j < 2; ++j)
#pragma unroll
                for (int e = 0; e < 8; ++e) {
                    int row = m0 + i * 16 + rb + e;   // global M = b*T + t
                    int col = n0 + j * 16 + lm;       // d_out
                    dst[(size_t)row * DD + col] = (__bf16)(acc[i][j][e] * mult);
                }
    } else {
#pragma unroll
        for (int i = 0; i < 2; ++i)
#pragma unroll
            for (int j = 0; j < 2; ++j)
#pragma unroll
                for (int e = 0; e < 8; ++e) {
                    int row = m0 + i * 16 + rb + e;
                    int col = n0 + j * 16 + lm;
                    int bb = row >> 11, t = row & (TT - 1);
                    vt[((size_t)(bb * DD + col)) * TT + t] =
                        (__bf16)acc[i][j][e];                 // [B, D, T]
                }
    }
}

// ---------------------------------------------------------------------------
// Kernel 2: causal flash attention.  One block (8 waves, 256 thr) per
// 16-row Q tile; wave w owns d-slice [w*128, w*128+128).  Key blocks of 32.
// Deterministic LDS reduction of partial S (no float atomics).
// ---------------------------------------------------------------------------
__global__ void flash_attn_kernel(const __bf16* __restrict__ qb,
                                  const __bf16* __restrict__ kbuf,
                                  const __bf16* __restrict__ vt,
                                  float* __restrict__ out) {
    __shared__ float Spart[8][16][32];                   // per-wave partial S
    __shared__ __align__(16) __bf16 Pld[16][32];         // P in bf16 for A-frags
    __shared__ float row_m[16], row_l[16], row_c[16];

    const int b    = blockIdx.y;
    const int q0   = blockIdx.x * 16;
    const int tid  = threadIdx.x;
    const int lane = tid & 31;
    const int w    = tid >> 5;
    const int lm   = lane & 15;
    const int hiA  = (lane >= 16) ? 8 : 0;
    const int hiB  = (lane >= 16) ? 16 : 0;
    const int rb   = (lane >= 16) ? 8 : 0;
    const int dbase = w * 128;

    if (tid < 16) { row_m[tid] = -__builtin_inff(); row_l[tid] = 0.0f; }

    v8f o[8];
#pragma unroll
    for (int n = 0; n < 8; ++n) o[n] = (v8f)(0.0f);
    __syncthreads();

    const int nkb = (q0 + 16 + 31) >> 5;                 // key blocks of 32
    for (int it = 0; it < nkb; ++it) {
        const int kb0 = it * 32;

        // ---- partial S = Q[:, slice] * K_block[:, slice]^T  (8 WMMA) ----
        v8f s[2];
        s[0] = (v8f)(0.0f); s[1] = (v8f)(0.0f);
#pragma unroll
        for (int dstep = 0; dstep < 4; ++dstep) {
            const int d = dbase + dstep * 32;
            const __bf16* pq = qb + ((size_t)(b * TT + q0 + lm)) * DD + d + hiA;
            v16bf A = make_frag(*(const bf8*)pq, *(const bf8*)(pq + 16));
#pragma unroll
            for (int nt = 0; nt < 2; ++nt) {
                const __bf16* pk =
                    kbuf + ((size_t)(b * TT + kb0 + nt * 16 + lm)) * DD + d + hiA;
                v16bf Bf = make_frag(*(const bf8*)pk, *(const bf8*)(pk + 16));
                s[nt] = wmma_bf16(A, Bf, s[nt]);
            }
        }
#pragma unroll
        for (int nt = 0; nt < 2; ++nt)
#pragma unroll
            for (int e = 0; e < 8; ++e)
                Spart[w][rb + e][nt * 16 + lm] = s[nt][e];
        __syncthreads();

        // ---- online softmax: thread t owns (row = t/16, 2 cols) ----
        // S already carries the 1/sqrt(D) scale (folded into Q).
        {
            const int r  = tid >> 4;
            const int h  = tid & 15;
            const int c0 = h * 2, c1 = c0 + 1;
            float s0 = 0.0f, s1 = 0.0f;
#pragma unroll
            for (int ww = 0; ww < 8; ++ww) {
                s0 += Spart[ww][r][c0];
                s1 += Spart[ww][r][c1];
            }
            if (kb0 + c0 > q0 + r) s0 = -__builtin_inff();
            if (kb0 + c1 > q0 + r) s1 = -__builtin_inff();

            float mx = fmaxf(s0, s1);
#pragma unroll
            for (int off = 1; off < 16; off <<= 1)
                mx = fmaxf(mx, __shfl_xor(mx, off, 16));

            float m_old = row_m[r];
            float m_new = fmaxf(m_old, mx);
            float cc = __expf(m_old - m_new);   // 0 on first touch (m_old=-inf)
            float p0 = __expf(s0 - m_new);
            float p1 = __expf(s1 - m_new);
            Pld[r][c0] = (__bf16)p0;
            Pld[r][c1] = (__bf16)p1;
            float ls = p0 + p1;
#pragma unroll
            for (int off = 1; off < 16; off <<= 1)
                ls += __shfl_xor(ls, off, 16);
            if (h == 0) {
                row_c[r] = cc;
                row_m[r] = m_new;
                row_l[r] = row_l[r] * cc + ls;
            }
        }
        __syncthreads();

        // ---- rescale O, then O += P * V_block[:, slice]  (8 WMMA) ----
#pragma unroll
        for (int e = 0; e < 8; ++e) {
            float ce = row_c[rb + e];
#pragma unroll
            for (int n = 0; n < 8; ++n) o[n][e] *= ce;
        }
        const __bf16* pp = &Pld[lm][hiA];
        v16bf Pa = make_frag(*(const bf8*)pp, *(const bf8*)(pp + 16));
#pragma unroll
        for (int n = 0; n < 8; ++n) {
            const __bf16* pv =
                vt + ((size_t)(b * DD + dbase + n * 16 + lm)) * TT + kb0 + hiA;
            v16bf Bv = make_frag(*(const bf8*)pv, *(const bf8*)(pv + 16));
            o[n] = wmma_bf16(Pa, Bv, o[n]);
        }
        // No barrier needed here: next iter's Spart writes are followed by a
        // barrier before anything reads them, and nothing below reads Pld.
    }

    // ---- epilogue: O / l, f32 store ----
#pragma unroll
    for (int e = 0; e < 8; ++e) {
        float inv = 1.0f / row_l[rb + e];
#pragma unroll
        for (int n = 0; n < 8; ++n) {
            int row = q0 + rb + e;
            int col = dbase + n * 16 + lm;
            out[((size_t)(b * TT + row)) * DD + col] = o[n][e] * inv;
        }
    }
}

// ---------------------------------------------------------------------------
// Host launcher.  Workspace layout (bytes, all 16B aligned):
//   xb  : 16 MB   Wqt/Wkt/Wvt : 3 x 2 MB   Qb : 16 MB   Kb : 16 MB   Vt : 16 MB
// ---------------------------------------------------------------------------
extern "C" void kernel_launch(void* const* d_in, const int* in_sizes, int n_in,
                              void* d_out, int out_size, void* d_ws, size_t ws_size,
                              hipStream_t stream) {
    const float* x  = (const float*)d_in[0];
    const float* Wq = (const float*)d_in[1];
    const float* Wk = (const float*)d_in[2];
    const float* Wv = (const float*)d_in[3];
    float* out = (float*)d_out;

    char* ws = (char*)d_ws;
    size_t off = 0;
    __bf16* xb  = (__bf16*)(ws + off); off += (size_t)MM * DD * 2;
    __bf16* wqt = (__bf16*)(ws + off); off += (size_t)DD * DD * 2;
    __bf16* wkt = (__bf16*)(ws + off); off += (size_t)DD * DD * 2;
    __bf16* wvt = (__bf16*)(ws + off); off += (size_t)DD * DD * 2;
    __bf16* qb  = (__bf16*)(ws + off); off += (size_t)MM * DD * 2;
    __bf16* kb  = (__bf16*)(ws + off); off += (size_t)MM * DD * 2;
    __bf16* vt  = (__bf16*)(ws + off); off += (size_t)MM * DD * 2;
    (void)ws_size; (void)in_sizes; (void)n_in; (void)out_size;

    // 0) precision/layout prep
    cvt_x_kernel<<<(MM * DD) / 256, 256, 0, stream>>>(x, xb, MM * DD);
    cvt_w_kernel<<<dim3((DD * DD) / 256, 3), 256, 0, stream>>>(
        Wq, Wk, Wv, wqt, wkt, wvt);

    // 1) Q,K,V projections (WMMA GEMM); Q pre-scaled by 1/sqrt(D)
    qkv_gemm_kernel<<<dim3(DD / 64, MM / 128, 3), 256, 0, stream>>>(
        xb, wqt, wkt, wvt, qb, kb, vt);

    // 2) causal flash attention (WMMA)
    flash_attn_kernel<<<dim3(TT / 16, BB), 256, 0, stream>>>(qb, kb, vt, out);
}
